// MultiHeadAttention_23252952940936
// MI455X (gfx1250) — compile-verified, matrix operand order checked by reading, not run on hardware
//
#include <hip/hip_runtime.h>

// ---------------- problem constants ----------------
#define BATCH   2
#define SEQ     2048
#define DMODEL  1024
#define NHEAD   16
#define HDIM    64
#define MTOT    (BATCH * SEQ)     // 4096 GEMM rows
#define NEGBIG  (-1e9f)

// ---------------- WMMA vector types ----------------
typedef __attribute__((ext_vector_type(16))) __bf16 v16bf;
typedef __attribute__((ext_vector_type(8)))  __bf16 bf16x8;
typedef __attribute__((ext_vector_type(8)))  float  v8f;

// f32 -> bf16 (round-to-nearest-even-ish)
static __device__ __forceinline__ unsigned short f2bf(float f) {
    unsigned int u = __float_as_uint(f);
    u += 0x7FFFu + ((u >> 16) & 1u);
    return (unsigned short)(u >> 16);
}

// Build a 16-element bf16 A/B fragment from two 16-byte chunks.
// CDNA5 16-bit A/B layout: lane half 0 holds K {0..7, 16..23},
// half 1 holds K {8..15, 24..31}; elements 0..7 = low chunk, 8..15 = high.
static __device__ __forceinline__ v16bf mk_frag(const unsigned short* lo,
                                                const unsigned short* hi) {
    bf16x8 a = *reinterpret_cast<const bf16x8*>(lo);
    bf16x8 b = *reinterpret_cast<const bf16x8*>(hi);
    v16bf r;
#pragma unroll
    for (int i = 0; i < 8; ++i) { r[i] = a[i]; r[i + 8] = b[i]; }
    return r;
}

static __device__ __forceinline__ v8f wmma_bf16(v16bf a, v16bf b, v8f c) {
    return __builtin_amdgcn_wmma_f32_16x16x32_bf16(
        false, a, false, b, (short)0, c, false, false);
}

// Async 16B copy global -> LDS (per active lane), tracked by ASYNCcnt.
static __device__ __forceinline__ void async_copy_b128(void* lds_dst,
                                                       const void* gsrc) {
    unsigned lds_off = (unsigned)(size_t)lds_dst;   // low 32 bits = LDS offset
    asm volatile("global_load_async_to_lds_b128 %0, %1, off"
                 :: "v"(lds_off), "v"(gsrc) : "memory");
}

static __device__ __forceinline__ void wait_async0() {
#if __has_builtin(__builtin_amdgcn_s_wait_asynccnt)
    __builtin_amdgcn_s_wait_asynccnt(0);
#else
    asm volatile("s_wait_asynccnt 0x0" ::: "memory");
#endif
}

// Partial wait: allow the 8 newest async ops (next block) to stay in flight.
static __device__ __forceinline__ void wait_async8() {
#if __has_builtin(__builtin_amdgcn_s_wait_asynccnt)
    __builtin_amdgcn_s_wait_asynccnt(8);
#else
    asm volatile("s_wait_asynccnt 0x8" ::: "memory");
#endif
}

// ---------------- kernel 0: f32 -> bf16 convert ----------------
__global__ void convert_bf16(const float* __restrict__ src,
                             unsigned short* __restrict__ dst, int n) {
    int i = blockIdx.x * blockDim.x + threadIdx.x;
    if (i < n) dst[i] = f2bf(src[i]);
}

// ---------------- kernel 1: projection GEMM ----------------
// out[m,n] = sum_k A[m,k] * W[n,k] + bias[n]   (torch Linear: x @ W^T + b)
// Wave tile 64(M) x 32(N): 4x2 WMMA accumulators, 8 WMMAs per K-step of 32.
// TR=0: out bf16 [MTOT, DMODEL] (Q, K);  TR=1: out bf16 [BATCH, DMODEL, SEQ] (V^T)
template <int TR>
__global__ void __launch_bounds__(256, 1)
proj_gemm(const unsigned short* __restrict__ A,
          const unsigned short* __restrict__ W,
          const float* __restrict__ bias,
          unsigned short* __restrict__ out) {
    const int lane = threadIdx.x & 31;
    const int wave = threadIdx.x >> 5;          // 0..7
    const int lm   = lane & 15;
    const int half = lane >> 4;
    const int waveM = wave & 1;                 // 2 waves along M
    const int waveN = wave >> 1;                // 4 waves along N
    const int mbase = blockIdx.y * 128 + waveM * 64;
    const int nbase = blockIdx.x * 128 + waveN * 32;

    v8f c[4][2] = {{{}, {}}, {{}, {}}, {{}, {}}, {{}, {}}};

    for (int kk = 0; kk < DMODEL; kk += 32) {
        v16bf af[4], bf[2];
#pragma unroll
        for (int i = 0; i < 4; ++i) {
            const unsigned short* p =
                A + (size_t)(mbase + i * 16 + lm) * DMODEL + kk + half * 8;
            af[i] = mk_frag(p, p + 16);
        }
#pragma unroll
        for (int j = 0; j < 2; ++j) {
            const unsigned short* p =
                W + (size_t)(nbase + j * 16 + lm) * DMODEL + kk + half * 8;
            bf[j] = mk_frag(p, p + 16);
        }
#pragma unroll
        for (int i = 0; i < 4; ++i)
#pragma unroll
            for (int j = 0; j < 2; ++j)
                c[i][j] = wmma_bf16(af[i], bf[j], c[i][j]);
    }

#pragma unroll
    for (int j = 0; j < 2; ++j) {
        const int n = nbase + j * 16 + lm;
        const float bj = bias[n];
#pragma unroll
        for (int i = 0; i < 4; ++i) {
#pragma unroll
            for (int r = 0; r < 8; ++r) {
                const int m = mbase + i * 16 + r + 8 * half;
                const float v = c[i][j][r] + bj;
                if (TR) {
                    const int bb = m >> 11;         // m / SEQ
                    const int l  = m & (SEQ - 1);
                    out[((size_t)bb * DMODEL + n) * SEQ + l] = f2bf(v);
                } else {
                    out[(size_t)m * DMODEL + n] = f2bf(v);
                }
            }
        }
    }
}

// ---------------- kernel 2: fused flash attention ----------------
// Q,K bf16 [B, L, D]; Vt bf16 [B, D, L]; mask int [B, L]; out f32 [B, L, D].
// 128 threads = 4 waves; each wave owns 16 query rows; key blocks of 64.
// K/V blocks double-buffered in LDS via async-to-LDS copies with partial
// ASYNCcnt waits so block i+1's copies overlap block i's WMMAs.
__global__ void __launch_bounds__(128, 1)
flash_attn(const unsigned short* __restrict__ Q,
           const unsigned short* __restrict__ K,
           const unsigned short* __restrict__ Vt,
           const int* __restrict__ mask,
           float* __restrict__ out) {
    const int tid  = threadIdx.x;
    const int lane = tid & 31;
    const int wave = tid >> 5;                  // 0..3
    const int lm   = lane & 15;
    const int half = lane >> 4;
    const int b    = blockIdx.y >> 4;
    const int h    = blockIdx.y & 15;
    const int qbase = blockIdx.x * 64 + wave * 16;

    __shared__ __align__(16) unsigned short Klds[2][64][72];  // [buf][key][d]
    __shared__ __align__(16) unsigned short Vlds[2][64][72];  // [buf][d][key]
    __shared__ __align__(16) unsigned short Plds[4][16][72];  // per-wave P tile

    // Issue the 8 async 16B copies (per lane) for one 64-key block.
    auto issue_copies = [&](int kb, int buf) {
#pragma unroll
        for (int it = 0; it < 4; ++it) {
            const int chunk = it * 128 + tid;       // 0..511
            const int row = chunk >> 3;             // 0..63
            const int c8  = (chunk & 7) * 8;        // element offset 0..56
            const unsigned short* gk =
                K + ((size_t)(b * SEQ + kb + row)) * DMODEL + h * HDIM + c8;
            async_copy_b128(&Klds[buf][row][c8], gk);
            const unsigned short* gv =
                Vt + ((size_t)(b * DMODEL + h * HDIM + row)) * SEQ + kb + c8;
            async_copy_b128(&Vlds[buf][row][c8], gv);
        }
    };

    // Q fragments (A-operand), loaded once from global
    v16bf aQ[2];
#pragma unroll
    for (int c = 0; c < 2; ++c) {
        const unsigned short* p =
            Q + ((size_t)(b * SEQ + qbase + lm)) * DMODEL + h * HDIM + c * 32 + half * 8;
        aQ[c] = mk_frag(p, p + 16);
    }

    float m_i[8], l_i[8];
#pragma unroll
    for (int r = 0; r < 8; ++r) { m_i[r] = -3.0e38f; l_i[r] = 0.0f; }
    v8f o[4] = {{}, {}, {}, {}};

    issue_copies(0, 0);   // prologue: stage first block

    int buf = 0;
    for (int kb = 0; kb < SEQ; kb += 64, buf ^= 1) {
        // additive key mask per N-tile (column = kb + t*16 + lm)
        float madd[4];
#pragma unroll
        for (int t = 0; t < 4; ++t)
            madd[t] = (mask[b * SEQ + kb + t * 16 + lm] == 0) ? NEGBIG : 0.0f;

        // overlap: issue next block's copies, then wait only for current block
        if (kb + 64 < SEQ) {
            issue_copies(kb + 64, buf ^ 1);
            wait_async8();          // in-order: <=8 outstanding => block kb done
        } else {
            wait_async0();
        }
        __syncthreads();

        // ---- S = Q @ K^T (contract over head dim, 2 steps of 32) ----
        v8f s[4] = {{}, {}, {}, {}};
#pragma unroll
        for (int c = 0; c < 2; ++c) {
#pragma unroll
            for (int t = 0; t < 4; ++t) {
                const unsigned short* p = &Klds[buf][t * 16 + lm][c * 32 + half * 8];
                s[t] = wmma_bf16(aQ[c], mk_frag(p, p + 16), s[t]);
            }
        }
#pragma unroll
        for (int t = 0; t < 4; ++t)
#pragma unroll
            for (int r = 0; r < 8; ++r)
                s[t][r] = s[t][r] * 0.125f + madd[t];   // 1/sqrt(64), then mask

        // ---- online softmax (rows striped across lanes 0..15 / 16..31) ----
#pragma unroll
        for (int r = 0; r < 8; ++r) {
            float pm = fmaxf(fmaxf(s[0][r], s[1][r]), fmaxf(s[2][r], s[3][r]));
#pragma unroll
            for (int off = 8; off > 0; off >>= 1)
                pm = fmaxf(pm, __shfl_xor(pm, off, 16));
            const float mn   = fmaxf(m_i[r], pm);
            const float corr = __expf(m_i[r] - mn);
            m_i[r] = mn;
            float ps = 0.0f;
#pragma unroll
            for (int t = 0; t < 4; ++t) {
                const float e = __expf(s[t][r] - mn);
                s[t][r] = e;
                ps += e;
            }
#pragma unroll
            for (int off = 8; off > 0; off >>= 1)
                ps += __shfl_xor(ps, off, 16);
            l_i[r] = l_i[r] * corr + ps;
#pragma unroll
            for (int t = 0; t < 4; ++t) o[t][r] *= corr;
        }

        // ---- P: C-layout -> A-layout via wave-private LDS tile ----
#pragma unroll
        for (int t = 0; t < 4; ++t)
#pragma unroll
            for (int r = 0; r < 8; ++r)
                Plds[wave][r + 8 * half][t * 16 + lm] = f2bf(s[t][r]);
        __syncthreads();

        // ---- O += P @ V (contract over 64 keys, 2 steps of 32) ----
#pragma unroll
        for (int c = 0; c < 2; ++c) {
            v16bf aP = mk_frag(&Plds[wave][lm][c * 32 + half * 8],
                               &Plds[wave][lm][c * 32 + 16 + half * 8]);
#pragma unroll
            for (int t = 0; t < 4; ++t) {
                const unsigned short* p = &Vlds[buf][t * 16 + lm][c * 32 + half * 8];
                o[t] = wmma_bf16(aP, mk_frag(p, p + 16), o[t]);
            }
        }
        __syncthreads();   // readers done before buf^1 gets overwritten next iter
    }

    // ---- normalize and store f32 output [B, L, H*Dh] ----
#pragma unroll
    for (int r = 0; r < 8; ++r) {
        const float inv = 1.0f / l_i[r];
        const int q = qbase + r + 8 * half;
#pragma unroll
        for (int t = 0; t < 4; ++t) {
            const int d = t * 16 + lm;
            out[((size_t)(b * SEQ + q)) * DMODEL + h * HDIM + d] = o[t][r] * inv;
        }
    }
}

// ---------------- host-side launch ----------------
extern "C" void kernel_launch(void* const* d_in, const int* in_sizes, int n_in,
                              void* d_out, int out_size, void* d_ws, size_t ws_size,
                              hipStream_t stream) {
    const float* feature = (const float*)d_in[0];
    const int*   mask    = (const int*)d_in[1];
    const float* W_Q     = (const float*)d_in[2];
    const float* b_Q     = (const float*)d_in[3];
    const float* W_K     = (const float*)d_in[4];
    const float* b_K     = (const float*)d_in[5];
    const float* W_V     = (const float*)d_in[6];
    const float* b_V     = (const float*)d_in[7];

    char* ws = (char*)d_ws;
    const size_t FEAT_N = (size_t)MTOT * DMODEL;      // 4M elems
    const size_t W_N    = (size_t)DMODEL * DMODEL;    // 1M elems
    unsigned short* feat_bf = (unsigned short*)(ws);
    unsigned short* Wq_bf   = (unsigned short*)(ws + 2 * FEAT_N);
    unsigned short* Wk_bf   = (unsigned short*)(ws + 2 * FEAT_N + 2 * W_N);
    unsigned short* Wv_bf   = (unsigned short*)(ws + 2 * FEAT_N + 4 * W_N);
    unsigned short* Qbf     = (unsigned short*)(ws + 2 * FEAT_N + 6 * W_N);
    unsigned short* Kbf     = (unsigned short*)(ws + 4 * FEAT_N + 6 * W_N);
    unsigned short* Vtbf    = (unsigned short*)(ws + 6 * FEAT_N + 6 * W_N);

    // 0) bf16 conversions
    convert_bf16<<<(int)((FEAT_N + 255) / 256), 256, 0, stream>>>(feature, feat_bf, (int)FEAT_N);
    convert_bf16<<<(int)((W_N + 255) / 256), 256, 0, stream>>>(W_Q, Wq_bf, (int)W_N);
    convert_bf16<<<(int)((W_N + 255) / 256), 256, 0, stream>>>(W_K, Wk_bf, (int)W_N);
    convert_bf16<<<(int)((W_N + 255) / 256), 256, 0, stream>>>(W_V, Wv_bf, (int)W_N);

    // 1) projections (WMMA GEMMs), macro tile 128x128
    dim3 pgrid(DMODEL / 128, MTOT / 128);
    proj_gemm<0><<<pgrid, 256, 0, stream>>>(feat_bf, Wq_bf, b_Q, Qbf);
    proj_gemm<0><<<pgrid, 256, 0, stream>>>(feat_bf, Wk_bf, b_K, Kbf);
    proj_gemm<1><<<pgrid, 256, 0, stream>>>(feat_bf, Wv_bf, b_V, Vtbf);

    // 2) fused flash attention
    dim3 agrid(SEQ / 64, BATCH * NHEAD);
    flash_attn<<<agrid, 128, 0, stream>>>(Qbf, Kbf, Vtbf, mask, (float*)d_out);
}
